// GBGraphConvModel_70995809402915
// MI455X (gfx1250) — matrix operand order repaired and checked by reading, not compiled, not run on hardware
//
#include <hip/hip_runtime.h>
#include <hip/hip_bf16.h>

typedef __attribute__((ext_vector_type(16))) _Float16 v16h;
typedef __attribute__((ext_vector_type(8)))  float    v8f;
typedef __attribute__((ext_vector_type(4)))  _Float16 v4h;

#define N_ATOMS 524288
#define BATCH   16384
#define BN_EPS  1e-3f

__constant__ int c_off[7] = {0, 8192, 73728, 204800, 401408, 499712, 516096};

// ---------------------------------------------------------------------------
// Pre-swizzle weights into WMMA B-fragment layout:
//   frag[w][kc][nh][lane][e] , e=0..15 halves (32B per lane)
//   K = kc*32 + (e&8)*2 + (lane>>4)*8 + (e&7) ;  N = (lane&15) + nh*16 ; zero-pad K >= Kreal
// ---------------------------------------------------------------------------
__global__ __launch_bounds__(256) void prep_kernel(
    const float* __restrict__ W1, const float* __restrict__ W2,
    const float* __restrict__ D1,
    _Float16* __restrict__ W1f, _Float16* __restrict__ W2f,
    _Float16* __restrict__ D1f)
{
    const int n1 = 13 * 3 * 2 * 512;   // gc1: 13 weights, K pad 96 (3 chunks), N=32 (2 halves)
    const int n2 = 13 * 1 * 2 * 512;   // gc2: K=32 (1 chunk)
    const int n3 = 4 * 512;            // d1 : K=32, N=64 (4 halves)
    int i = blockIdx.x * 256 + threadIdx.x;
    if (i < n1) {
        int e = i & 15, lane = (i >> 4) & 31, nh = (i >> 9) & 1;
        int kc = (i >> 10) % 3, w = i / 3072;
        int K = kc * 32 + ((e & 8) << 1) + (lane >> 4) * 8 + (e & 7);
        int N = (lane & 15) + nh * 16;
        W1f[i] = (K < 75) ? (_Float16)W1[((size_t)w * 75 + K) * 32 + N] : (_Float16)0.f;
    } else if (i < n1 + n2) {
        int j = i - n1;
        int e = j & 15, lane = (j >> 4) & 31, nh = (j >> 9) & 1, w = j >> 10;
        int K = ((e & 8) << 1) + (lane >> 4) * 8 + (e & 7);
        int N = (lane & 15) + nh * 16;
        W2f[j] = (_Float16)W2[((size_t)w * 32 + K) * 32 + N];
    } else if (i < n1 + n2 + n3) {
        int j = i - n1 - n2;
        int e = j & 15, lane = (j >> 4) & 31, nh = j >> 9;
        int K = ((e & 8) << 1) + (lane >> 4) * 8 + (e & 7);
        int N = (lane & 15) + nh * 16;
        D1f[j] = (_Float16)D1[(size_t)K * 64 + N];
    }
}

// ---------------------------------------------------------------------------
// Graph conv + bias + tanh + BN(32ch), one wave handles one 16-row tile.
// T = float (layer 1, K=75, NK=3) or _Float16 (layer 2, K=32, NK=1).
// ---------------------------------------------------------------------------
template <typename T, int K, int NK>
__global__ __launch_bounds__(32) void conv_kernel(
    const T* __restrict__ X, const _Float16* __restrict__ Wfrag,
    const float* __restrict__ gcb,
    const float* __restrict__ bn_g, const float* __restrict__ bn_b,
    const float* __restrict__ bn_m, const float* __restrict__ bn_v,
    _Float16* __restrict__ Y,
    const int* __restrict__ a1, const int* __restrict__ a2,
    const int* __restrict__ a3, const int* __restrict__ a4,
    const int* __restrict__ a5, const int* __restrict__ a6)
{
    __shared__ float sm_self[16][K + 2];
    __shared__ float sm_rel [16][K + 2];
    __shared__ int   sidx[16 * 6];

    const int lane = threadIdx.x;
    const int row0 = blockIdx.x * 16;

    int d = 0, segStart = 0;
    #pragma unroll
    for (int i = 1; i <= 6; ++i)
        if (row0 >= c_off[i]) { d = i; segStart = c_off[i]; }

    const int* adjp = (d == 1) ? a1 : (d == 2) ? a2 : (d == 3) ? a3
                    : (d == 4) ? a4 : (d == 5) ? a5 : a6;

    if (d > 0)
        for (int t = lane; t < 16 * d; t += 32)
            sidx[t] = adjp[(size_t)(row0 - segStart) * d + t];
    __syncthreads();

    for (int t = lane; t < 16 * K; t += 32) {
        int m = t / K, c = t - m * K;
        sm_self[m][c] = (float)X[(size_t)(row0 + m) * K + c];
        if (d > 0) {
            float r = 0.f;
            for (int j = 0; j < d; ++j)
                r += (float)X[(size_t)sidx[m * d + j] * K + c];
            sm_rel[m][c] = r;
        }
    }
    __syncthreads();

    const int half = lane >> 4;
    const int mrow = lane & 15;
    v8f acc0 = {};
    v8f acc1 = {};
    const int wsel_self = (d == 0) ? 12 : (2 * (d - 1) + 1);
    const int wsel_rel  = 2 * (d - 1);
    const _Float16* wb_s = Wfrag + (size_t)wsel_self * NK * 1024;
    const _Float16* wb_r = Wfrag + (size_t)wsel_rel  * NK * 1024;

    #pragma unroll
    for (int kc = 0; kc < NK; ++kc) {
        v16h a_s;
        #pragma unroll
        for (int e = 0; e < 16; ++e) {
            int v = e >> 1, lo = e & 1;
            int k = kc * 32 + ((v & 4) ? 16 : 0) + half * 8 + (v & 3) * 2 + lo;
            a_s[e] = (k < K) ? (_Float16)sm_self[mrow][k] : (_Float16)0.f;
        }
        const _Float16* ps = wb_s + (size_t)kc * 1024 + lane * 16;
        v16h b0 = *(const v16h*)(ps);
        v16h b1 = *(const v16h*)(ps + 512);
        acc0 = __builtin_amdgcn_wmma_f32_16x16x32_f16(false, a_s, false, b0,
                                                      (short)0, acc0, false, false);
        acc1 = __builtin_amdgcn_wmma_f32_16x16x32_f16(false, a_s, false, b1,
                                                      (short)0, acc1, false, false);
        if (d > 0) {
            v16h a_r;
            #pragma unroll
            for (int e = 0; e < 16; ++e) {
                int v = e >> 1, lo = e & 1;
                int k = kc * 32 + ((v & 4) ? 16 : 0) + half * 8 + (v & 3) * 2 + lo;
                a_r[e] = (k < K) ? (_Float16)sm_rel[mrow][k] : (_Float16)0.f;
            }
            const _Float16* pr = wb_r + (size_t)kc * 1024 + lane * 16;
            v16h c0 = *(const v16h*)(pr);
            v16h c1 = *(const v16h*)(pr + 512);
            acc0 = __builtin_amdgcn_wmma_f32_16x16x32_f16(false, a_r, false, c0,
                                                          (short)0, acc0, false, false);
            acc1 = __builtin_amdgcn_wmma_f32_16x16x32_f16(false, a_r, false, c1,
                                                          (short)0, acc1, false, false);
        }
    }

    // epilogue: bias + tanh + batchnorm, store f16
    #pragma unroll
    for (int nh = 0; nh < 2; ++nh) {
        int N = mrow + nh * 16;
        float bias = (d == 0) ? gcb[12 * 32 + N]
                              : gcb[(2 * (d - 1)) * 32 + N] + gcb[(2 * (d - 1) + 1) * 32 + N];
        float gam = bn_g[N], bet = bn_b[N], mu = bn_m[N];
        float rs = rsqrtf(bn_v[N] + BN_EPS);
        v8f acc = nh ? acc1 : acc0;
        #pragma unroll
        for (int v = 0; v < 8; ++v) {
            int M = v + half * 8;
            float t = tanhf(acc[v] + bias);
            float y = gam * (t - mu) * rs + bet;
            Y[(size_t)(row0 + M) * 32 + N] = (_Float16)y;
        }
    }
}

// ---------------------------------------------------------------------------
// Graph pool: out[i] = max(self, max over neighbors); degree-0 passthrough.
// One thread handles 4 channels (8B vector) of one atom.
// ---------------------------------------------------------------------------
__global__ __launch_bounds__(256) void pool_kernel(
    const _Float16* __restrict__ Hin, _Float16* __restrict__ Hout,
    const int* __restrict__ a1, const int* __restrict__ a2,
    const int* __restrict__ a3, const int* __restrict__ a4,
    const int* __restrict__ a5, const int* __restrict__ a6)
{
    int gid = blockIdx.x * 256 + threadIdx.x;   // N_ATOMS * 8
    int atom = gid >> 3, q = gid & 7;

    int d = 0, segStart = 0;
    #pragma unroll
    for (int i = 1; i <= 6; ++i)
        if (atom >= c_off[i]) { d = i; segStart = c_off[i]; }

    const v4h* in = (const v4h*)Hin;
    v4h val = in[(size_t)atom * 8 + q];
    if (d > 0) {
        const int* adjp = (d == 1) ? a1 : (d == 2) ? a2 : (d == 3) ? a3
                        : (d == 4) ? a4 : (d == 5) ? a5 : a6;
        size_t base = (size_t)(atom - segStart) * d;
        float m0 = (float)val[0], m1 = (float)val[1];
        float m2 = (float)val[2], m3 = (float)val[3];
        for (int j = 0; j < d; ++j) {
            int nb = adjp[base + j];
            v4h nv = in[(size_t)nb * 8 + q];
            m0 = fmaxf(m0, (float)nv[0]); m1 = fmaxf(m1, (float)nv[1]);
            m2 = fmaxf(m2, (float)nv[2]); m3 = fmaxf(m3, (float)nv[3]);
        }
        val[0] = (_Float16)m0; val[1] = (_Float16)m1;
        val[2] = (_Float16)m2; val[3] = (_Float16)m3;
    }
    ((v4h*)Hout)[(size_t)atom * 8 + q] = val;
}

// ---------------------------------------------------------------------------
// Per-molecule tail: dense 32->64 (+tanh +BN3) over the 32 atoms of molecule m
// (atoms m + k*16384), segment sum/max, tanh, d2 dot, d3 tail. One wave/mol.
// ---------------------------------------------------------------------------
__global__ __launch_bounds__(32) void mol_kernel(
    const _Float16* __restrict__ P, const _Float16* __restrict__ D1f,
    const float* __restrict__ d1b,
    const float* __restrict__ g3, const float* __restrict__ b3,
    const float* __restrict__ m3, const float* __restrict__ v3,
    const float* __restrict__ d2W, const float* __restrict__ d2b,
    const float* __restrict__ d3W, const float* __restrict__ d3b,
    const float* __restrict__ xadd, float* __restrict__ out)
{
    const int mol  = blockIdx.x;
    const int lane = threadIdx.x;
    const int half = lane >> 4;
    const int mrow = lane & 15;

    // A fragments for the two 16-row M tiles (K = 32, f16 already)
    v16h a0, a1;
    #pragma unroll
    for (int e = 0; e < 16; ++e) {
        int v = e >> 1, lo = e & 1;
        int k = ((v & 4) ? 16 : 0) + half * 8 + (v & 3) * 2 + lo;
        size_t r0 = (size_t)(mol + mrow * BATCH) * 32 + k;
        size_t r1 = (size_t)(mol + (mrow + 16) * BATCH) * 32 + k;
        a0[e] = P[r0];
        a1[e] = P[r1];
    }

    v8f zero = {};
    v8f acc[2][4];
    #pragma unroll
    for (int nh = 0; nh < 4; ++nh) {
        v16h b = *(const v16h*)(D1f + (size_t)nh * 512 + lane * 16);
        acc[0][nh] = __builtin_amdgcn_wmma_f32_16x16x32_f16(false, a0, false, b,
                                                            (short)0, zero, false, false);
        acc[1][nh] = __builtin_amdgcn_wmma_f32_16x16x32_f16(false, a1, false, b,
                                                            (short)0, zero, false, false);
    }

    float contrib = 0.f;
    #pragma unroll
    for (int nh = 0; nh < 4; ++nh) {
        int N = mrow + nh * 16;
        float bias = d1b[N];
        float gam = g3[N], bet = b3[N], mu = m3[N];
        float rs = rsqrtf(v3[N] + BN_EPS);
        float s = 0.f, mx = -3.402823466e38f;
        #pragma unroll
        for (int mt = 0; mt < 2; ++mt)
            #pragma unroll
            for (int v = 0; v < 8; ++v) {
                float t = tanhf(acc[mt][nh][v] + bias);
                float y = gam * (t - mu) * rs + bet;
                s += y;
                mx = fmaxf(mx, y);
            }
        // merge the two 16-lane halves (lane ^ 16 covers the other 8 M rows)
        s  += __shfl_xor(s, 16, 32);
        mx  = fmaxf(mx, __shfl_xor(mx, 16, 32));
        contrib += tanhf(s) * d2W[N] + tanhf(mx) * d2W[64 + N];
    }
    if (lane >= 16) contrib = 0.f;          // lanes 16..31 are duplicates
    #pragma unroll
    for (int off = 8; off >= 1; off >>= 1)
        contrib += __shfl_xor(contrib, off, 32);

    if (lane == 0) {
        float model_var = contrib + d2b[0];
        float ans = model_var * d3W[0];
        for (int j = 0; j < 15; ++j)
            ans += xadd[(size_t)mol * 15 + j] * d3W[j + 1];
        ans += d3b[0];
        out[mol] = ans;
    }
}

// ---------------------------------------------------------------------------
extern "C" void kernel_launch(void* const* d_in, const int* in_sizes, int n_in,
                              void* d_out, int out_size, void* d_ws, size_t ws_size,
                              hipStream_t stream)
{
    (void)in_sizes; (void)n_in; (void)out_size; (void)ws_size;
    const float* atomf = (const float*)d_in[0];
    const int* a1 = (const int*)d_in[2];
    const int* a2 = (const int*)d_in[3];
    const int* a3 = (const int*)d_in[4];
    const int* a4 = (const int*)d_in[5];
    const int* a5 = (const int*)d_in[6];
    const int* a6 = (const int*)d_in[7];
    const float* W1  = (const float*)d_in[8];
    const float* b1  = (const float*)d_in[9];
    const float* W2  = (const float*)d_in[10];
    const float* b2  = (const float*)d_in[11];
    const float* g1  = (const float*)d_in[12];
    const float* be1 = (const float*)d_in[13];
    const float* mu1 = (const float*)d_in[14];
    const float* vr1 = (const float*)d_in[15];
    const float* g3  = (const float*)d_in[16];
    const float* be3 = (const float*)d_in[17];
    const float* mu3 = (const float*)d_in[18];
    const float* vr3 = (const float*)d_in[19];
    const float* d1W = (const float*)d_in[20];
    const float* d1b = (const float*)d_in[21];
    const float* d2W = (const float*)d_in[22];
    const float* d2b = (const float*)d_in[23];
    const float* d3W = (const float*)d_in[24];
    const float* d3b = (const float*)d_in[25];
    const float* xadd = (const float*)d_in[26];
    float* out = (float*)d_out;

    // workspace: 2 x (N_ATOMS*32) f16 activation buffers (32MB each) + frags
    _Float16* bufA = (_Float16*)d_ws;
    _Float16* bufB = bufA + (size_t)N_ATOMS * 32;
    _Float16* w1f  = bufB + (size_t)N_ATOMS * 32;
    _Float16* w2f  = w1f + 13 * 3 * 2 * 512;
    _Float16* d1f  = w2f + 13 * 1 * 2 * 512;

    prep_kernel<<<216, 256, 0, stream>>>(W1, W2, d1W, w1f, w2f, d1f);

    conv_kernel<float, 75, 3><<<N_ATOMS / 16, 32, 0, stream>>>(
        atomf, w1f, b1, g1, be1, mu1, vr1, bufA, a1, a2, a3, a4, a5, a6);

    pool_kernel<<<(N_ATOMS * 8) / 256, 256, 0, stream>>>(
        bufA, bufB, a1, a2, a3, a4, a5, a6);

    conv_kernel<_Float16, 32, 1><<<N_ATOMS / 16, 32, 0, stream>>>(
        bufB, w2f, b2, g1, be1, mu1, vr1, bufA, a1, a2, a3, a4, a5, a6);

    pool_kernel<<<(N_ATOMS * 8) / 256, 256, 0, stream>>>(
        bufA, bufB, a1, a2, a3, a4, a5, a6);

    mol_kernel<<<BATCH, 32, 0, stream>>>(
        bufB, d1f, d1b, g3, be3, mu3, vr3, d2W, d2b, d3W, d3b, xadd, out);
}
